// SIAEncoder_86328842650010
// MI455X (gfx1250) — matrix operand order repaired and checked by reading, not compiled, not run on hardware
//
#include <hip/hip_runtime.h>
#include <math.h>
#include <stdint.h>

#define PADTOK 0
#define NHASH 8

typedef __attribute__((ext_vector_type(16))) _Float16 v16h;
typedef __attribute__((ext_vector_type(8)))  _Float16 v8h;
typedef __attribute__((ext_vector_type(8)))  float    v8f;

// ---------------------------------------------------------------------------
// WMMA fragment layouts (CDNA5 ISA §7.12.2, wave32):
//  A 16x32 f16: lane l -> m=l&15, half=l>>4; elem e -> k = (e>>3)*16 + half*8 + (e&7)
//    => per lane two contiguous 8-f16 runs: [half*8,+8) and [16+half*8,+8)
//  B 32x16 f16: lane l -> n=l&15, half=l>>4; elem e -> k = (e>>3)*16 + half*8 + (e&7), as for A
//    => same two 8-f16 runs as A; k_lsh_attn/k_gru use k = half*16 + e for both A and B
//  C 16x16 f32: lane l -> n=l&15; vgpr r -> m = (l>>4)*8 + r
// ---------------------------------------------------------------------------
__device__ __forceinline__ v8f wmma_f32_f16(v16h a, v16h b, v8f c) {
  return __builtin_amdgcn_wmma_f32_16x16x32_f16(false, a, false, b, (short)0, c,
                                                false, false);
}

__device__ __forceinline__ v16h pack16(float4 a, float4 b, float4 c, float4 d) {
  v16h r;
  r[0] = (_Float16)a.x;  r[1] = (_Float16)a.y;  r[2] = (_Float16)a.z;  r[3] = (_Float16)a.w;
  r[4] = (_Float16)b.x;  r[5] = (_Float16)b.y;  r[6] = (_Float16)b.z;  r[7] = (_Float16)b.w;
  r[8] = (_Float16)c.x;  r[9] = (_Float16)c.y;  r[10] = (_Float16)c.z; r[11] = (_Float16)c.w;
  r[12] = (_Float16)d.x; r[13] = (_Float16)d.y; r[14] = (_Float16)d.z; r[15] = (_Float16)d.w;
  return r;
}
__device__ __forceinline__ v16h pack16s(float4 a, float4 b, float4 c, float4 d, float s) {
  v16h r;
  r[0] = (_Float16)(a.x * s);  r[1] = (_Float16)(a.y * s);
  r[2] = (_Float16)(a.z * s);  r[3] = (_Float16)(a.w * s);
  r[4] = (_Float16)(b.x * s);  r[5] = (_Float16)(b.y * s);
  r[6] = (_Float16)(b.z * s);  r[7] = (_Float16)(b.w * s);
  r[8] = (_Float16)(c.x * s);  r[9] = (_Float16)(c.y * s);
  r[10] = (_Float16)(c.z * s); r[11] = (_Float16)(c.w * s);
  r[12] = (_Float16)(d.x * s); r[13] = (_Float16)(d.y * s);
  r[14] = (_Float16)(d.z * s); r[15] = (_Float16)(d.w * s);
  return r;
}
__device__ __forceinline__ v8h pack8(float4 a, float4 b) {
  v8h r;
  r[0] = (_Float16)a.x; r[1] = (_Float16)a.y; r[2] = (_Float16)a.z; r[3] = (_Float16)a.w;
  r[4] = (_Float16)b.x; r[5] = (_Float16)b.y; r[6] = (_Float16)b.z; r[7] = (_Float16)b.w;
  return r;
}
#define CAT16(lo, hi) \
  __builtin_shufflevector(lo, hi, 0, 1, 2, 3, 4, 5, 6, 7, 8, 9, 10, 11, 12, 13, 14, 15)

template <typename F>
__device__ __forceinline__ void store_c_frag(const v8f& c, F f) {
  const int l = threadIdx.x & 31;
  const int n = l & 15, mh = (l >> 4) << 3;
#pragma unroll
  for (int r = 0; r < 8; ++r) f(mh + r, n, c[r]);
}

__device__ __forceinline__ float sigmf(float x) { return 1.0f / (1.0f + expf(-x)); }

// ---------------------------------------------------------------------------
// Embedding gather (D hardcoded 512)
// ---------------------------------------------------------------------------
__global__ void k_embed(const int* __restrict__ ids, const float* __restrict__ emb,
                        float* __restrict__ outp, long total) {
  long idx = (long)blockIdx.x * blockDim.x + threadIdx.x;
  if (idx >= total) return;
  long r = idx >> 9;
  int cd = (int)(idx & 511);
  outp[idx] = emb[(long)ids[r] * 512 + cd];
}

// ---------------------------------------------------------------------------
// Deterministic N(0,1) fill (hash + Box-Muller) for LSH rotation matrices
// ---------------------------------------------------------------------------
__device__ __forceinline__ unsigned hash_u32(unsigned x) {
  x ^= x >> 16; x *= 0x7feb352dU; x ^= x >> 15; x *= 0x846ca68bU; x ^= x >> 16;
  return x;
}
__global__ void k_randn(float* __restrict__ outp, long n, unsigned seed) {
  long i = (long)blockIdx.x * blockDim.x + threadIdx.x;
  if (i >= n) return;
  unsigned h1 = hash_u32(seed ^ (unsigned)(2 * i + 1));
  unsigned h2 = hash_u32((seed * 0x9e3779b9U) ^ (unsigned)(2 * i + 2));
  float u1 = ((float)h1 + 1.0f) * (1.0f / 4294967296.0f);
  float u2 = (float)h2 * (1.0f / 4294967296.0f);
  outp[i] = sqrtf(-2.0f * logf(u1)) * cosf(6.2831853f * u2);
}

// ---------------------------------------------------------------------------
// WMMA GEMM, LDS-staged. REQUIRES M%64==0, N%64==0, K%32==0 (true for all
// call sites). Block 256 thr = 8 waves; block tile 64x64, each wave owns a
// 32x16 strip = 2 WMMA tiles. A and B^T staged in LDS each 32-k step with
// contiguous b128 global loads; fragments read back as b128 LDS loads.
//   transB:   B is [N,K] row-major (ld=ldb), else [K,N] (ld=ldb)
//   splitOut: write C as head-split [B,8,2048,64]
//   headsA:   read A from head-split [B,8,2048,64]
// ---------------------------------------------------------------------------
__global__ void k_gemm(const float* __restrict__ A, const float* __restrict__ Bm,
                       const float* __restrict__ bias, float* __restrict__ C,
                       int M, int N, int K, int transB, int ldb,
                       int splitOut, int headsA) {
  __shared__ _Float16 As[64 * 32];   // As[row][k]
  __shared__ _Float16 Bst[64 * 32];  // Bst[n][k]  (B transposed)
  const int tid = threadIdx.x;
  const int wave = tid >> 5;
  const int wr = wave >> 2, wc = wave & 3;
  const int l = tid & 31, lm = l & 15, lh = l >> 4;
  const long rowBase = (long)blockIdx.y * 64;
  const long colBase = (long)blockIdx.x * 64;
  v8f accs[2] = {{}, {}};

  const int arow = tid >> 2, akq = (tid & 3) << 3;  // A stage coords
  const int tbn = tid >> 2, tbk = (tid & 3) << 3;   // B stage (transB)
  const int nbk = tid >> 3, nbn = (tid & 7) << 3;   // B stage (normal)

  for (int k0 = 0; k0 < K; k0 += 32) {
    // ---- stage A tile ----
    {
      long gi = rowBase + arow;
      v8h h;
      if (!headsA) {
        if (k0 + 32 < K) __builtin_prefetch(A + gi * K + k0 + 32 + akq, 0, 0);
        const float4* p = (const float4*)(A + gi * (long)K + k0 + akq);
        h = pack8(p[0], p[1]);
      } else {  // gather from [B,8,2048,64]
        long b = gi >> 11, t = gi & 2047;
#pragma unroll
        for (int j = 0; j < 8; ++j) {
          long kk = k0 + akq + j;
          h[j] = (_Float16)A[(((b << 3) + (kk >> 6)) * 2048 + t) * 64 + (kk & 63)];
        }
      }
      *(v8h*)&As[arow * 32 + akq] = h;
    }
    // ---- stage B tile (transposed into Bst[n][k]) ----
    if (transB) {
      const float4* p = (const float4*)(Bm + (long)(colBase + tbn) * ldb + k0 + tbk);
      *(v8h*)&Bst[tbn * 32 + tbk] = pack8(p[0], p[1]);
    } else {
      const float4* p = (const float4*)(Bm + (long)(k0 + nbk) * ldb + colBase + nbn);
      float4 x0 = p[0], x1 = p[1];
      Bst[(nbn + 0) * 32 + nbk] = (_Float16)x0.x;
      Bst[(nbn + 1) * 32 + nbk] = (_Float16)x0.y;
      Bst[(nbn + 2) * 32 + nbk] = (_Float16)x0.z;
      Bst[(nbn + 3) * 32 + nbk] = (_Float16)x0.w;
      Bst[(nbn + 4) * 32 + nbk] = (_Float16)x1.x;
      Bst[(nbn + 5) * 32 + nbk] = (_Float16)x1.y;
      Bst[(nbn + 6) * 32 + nbk] = (_Float16)x1.z;
      Bst[(nbn + 7) * 32 + nbk] = (_Float16)x1.w;
    }
    __syncthreads();
    // ---- compute: 2 WMMA tiles per wave ----
    const v8h b0 = *(const v8h*)&Bst[(wc * 16 + lm) * 32 + lh * 8];
    const v8h b1 = *(const v8h*)&Bst[(wc * 16 + lm) * 32 + 16 + lh * 8];
    const v16h bf = CAT16(b0, b1);
#pragma unroll
    for (int sub = 0; sub < 2; ++sub) {
      const int r = (wr * 32 + sub * 16 + lm) * 32;
      const v8h a0 = *(const v8h*)&As[r + lh * 8];
      const v8h a1 = *(const v8h*)&As[r + 16 + lh * 8];
      accs[sub] = wmma_f32_f16(CAT16(a0, a1), bf, accs[sub]);
    }
    __syncthreads();
  }
  // ---- epilogue ----
#pragma unroll
  for (int sub = 0; sub < 2; ++sub) {
    store_c_frag(accs[sub], [&](int m, int n, float v) {
      long i = rowBase + wr * 32 + sub * 16 + m;
      long j = colBase + wc * 16 + n;
      float o = v + (bias ? bias[j] : 0.0f);
      if (splitOut) {
        long b = i >> 11, t = i & 2047;
        C[(((b << 3) + (j >> 6)) * 2048 + t) * 64 + (j & 63)] = o;
      } else {
        C[i * (long)N + j] = o;
      }
    });
  }
}

// ---------------------------------------------------------------------------
// Bucket assignment: argmax over concat(rot, -rot); Tt = 1<<ttShift
// ---------------------------------------------------------------------------
__global__ void k_buckets(const float* __restrict__ rotated, int* __restrict__ buckets,
                          long total, int ttShift, int R) {
  long idx = (long)blockIdx.x * blockDim.x + threadIdx.x;
  if (idx >= total) return;
  const int Tt = 1 << ttShift;
  int pos = (int)(idx & (Tt - 1));
  long r1 = idx >> ttShift;
  int h = (int)(r1 & 7);
  long b = r1 >> 3;
  const float* rr = rotated + (b * Tt + pos) * (NHASH * (long)R) + h * R;
  float best = -3.4e38f;
  int bi = 0;
  for (int r = 0; r < 2 * R; ++r) {
    float val = (r < R) ? rr[r] : -rr[r - R];
    if (val > best) { best = val; bi = r; }
  }
  buckets[(b * NHASH + h) * Tt + pos] = bi + h * (2 * R);
}

// ---------------------------------------------------------------------------
// Stable counting sort per batch row (== argsort(bucket*t+pos): bucket ids are
// disjoint across hash rounds, so position order within a bucket is stable).
// ---------------------------------------------------------------------------
__global__ void k_sort(const int* __restrict__ buckets, int* __restrict__ sticker,
                       int* __restrict__ undo, int* __restrict__ counts,
                       int NB, int n, int nbins) {
  int b = blockIdx.x * blockDim.x + threadIdx.x;
  if (b >= NB) return;
  int* cnt = counts + (long)b * nbins;
  const int* bk = buckets + (long)b * n;
  for (int k = 0; k < nbins; ++k) cnt[k] = 0;
  for (int i = 0; i < n; ++i) cnt[bk[i]]++;
  int acc = 0;
  for (int k = 0; k < nbins; ++k) { int t = cnt[k]; cnt[k] = acc; acc += t; }
  for (int i = 0; i < n; ++i) {
    int r = cnt[bk[i]]++;
    sticker[(long)b * n + r] = i;
    undo[(long)b * n + i] = r;
  }
}

// ---------------------------------------------------------------------------
// LSH chunked attention, one block (256 thr) per (batch, chunk).
// dots (32x64) via WMMA (vectorized b128 gathers), LDS logsumexp softmax,
// p@v via WMMA. ttMask=Tt-1, ncMask=nchunks-1, hpbShift=log2(heads/batch).
// d, dv are multiples of 32/16.
// ---------------------------------------------------------------------------
__global__ void k_lsh_attn(const float* __restrict__ qk, const float* __restrict__ v,
                           const int* __restrict__ sticker, const int* __restrict__ xs,
                           float* __restrict__ so, float* __restrict__ sl,
                           int Tt, int ttMask, int ncMask, int hpbShift,
                           int d, int dv, int nSorted, float scale) {
  __shared__ int posq[32];
  __shared__ int posk[64];
  __shared__ int mk[64];
  __shared__ float invn[64];
  __shared__ float p[32][64];
  const int bb = blockIdx.y;
  const int c = blockIdx.x;
  const int tid = threadIdx.x;
  const long sbase = (long)bb * nSorted;
  const int l = tid & 31, lm = l & 15, lh = l >> 4;

  if (tid < 32) posq[tid] = sticker[sbase + c * 32 + tid] & ttMask;
  if (tid < 64) {
    int cc = (tid < 32) ? c : ((c - 1) & ncMask);
    int pos = sticker[sbase + cc * 32 + (tid & 31)] & ttMask;
    posk[tid] = pos;
    mk[tid] = (xs[(bb >> hpbShift) * 1024 + (pos & 1023)] == PADTOK) ? 1 : 0;
  }
  __syncthreads();
  if (tid < 64) {  // keys = normalized queries
    const float* row = qk + ((long)bb * Tt + posk[tid]) * d;
    float s = 0.f;
    for (int k = 0; k < d; k += 4) {
      float4 x = *(const float4*)(row + k);
      s += x.x * x.x + x.y * x.y + x.z * x.z + x.w * x.w;
    }
    invn[tid] = 1.0f / (sqrtf(s) + 1e-9f);
  }
  __syncthreads();

  const int wave = tid >> 5;
  {  // dots = q . k_hat^T, 2x4 tiles over 8 waves
    const int mi = wave >> 2, ni = wave & 3;
    const int j = ni * 16 + lm;
    const float sInv = invn[j];
    const long qrow = (long)bb * Tt + posq[mi * 16 + lm];
    const long krow = (long)bb * Tt + posk[j];
    v8f acc = {};
    for (int k0 = 0; k0 < d; k0 += 32) {
      const float* ab = qk + qrow * d + k0;
      const float4* a0 = (const float4*)(ab + lh * 16);
      const float4* a1 = (const float4*)(ab + lh * 16 + 8);
      const float4* bp = (const float4*)(qk + krow * d + k0 + lh * 16);
      acc = wmma_f32_f16(pack16(a0[0], a0[1], a1[0], a1[1]),
                         pack16s(bp[0], bp[1], bp[2], bp[3], sInv), acc);
    }
    store_c_frag(acc, [&](int m, int n, float val) {
      int i = mi * 16 + m, jj = ni * 16 + n;
      val *= scale;
      if (posq[i] == posk[jj]) val = -5.0e4f;  // self-attention penalty
      if (mk[jj]) val = -1.0e9f;               // padded keys
      p[i][jj] = val;
    });
  }
  __syncthreads();
  if (tid < 32) {  // per-row logsumexp, normalize in place
    float mx = -3.4e38f;
    for (int j2 = 0; j2 < 64; ++j2) mx = fmaxf(mx, p[tid][j2]);
    float s = 0.f;
    for (int j2 = 0; j2 < 64; ++j2) s += expf(p[tid][j2] - mx);
    float lse = mx + logf(s);
    for (int j2 = 0; j2 < 64; ++j2) p[tid][j2] = expf(p[tid][j2] - lse);
    sl[sbase + c * 32 + tid] = lse;
  }
  __syncthreads();

  const int totalTiles = 2 * (dv >> 4);  // o = p @ v
  for (int tt = wave; tt < totalTiles; tt += 8) {
    const int mi = tt & 1, ni = tt >> 1;
    const float* prow = &p[mi * 16 + lm][0];
    v8f acc = {};
#pragma unroll
    for (int k0 = 0; k0 < 64; k0 += 32) {
      const float4* a0 = (const float4*)(prow + k0 + lh * 16);
      const float4* a1 = (const float4*)(prow + k0 + lh * 16 + 8);
      v16h bf;
#pragma unroll
      for (int e = 0; e < 16; ++e) {
        int j = k0 + lh * 16 + e;
        bf[e] = (_Float16)v[((long)bb * Tt + posk[j]) * dv + (ni * 16 + lm)];
      }
      acc = wmma_f32_f16(pack16(a0[0], a0[1], a1[0], a1[1]), bf, acc);
    }
    store_c_frag(acc, [&](int m, int n, float val) {
      so[(sbase + c * 32 + mi * 16 + m) * (long)dv + ni * 16 + n] = val;
    });
  }
}

// ---------------------------------------------------------------------------
// Unsort + combine hash rounds: out[b,pos,:] = sum_h softmax_h(lse) * so[undo]
// ---------------------------------------------------------------------------
__global__ void k_combine(const float* __restrict__ so, const float* __restrict__ sl,
                          const int* __restrict__ undo, float* __restrict__ outp,
                          int Tt, int dv, int nSorted, long outBatchStride) {
  const int bb = blockIdx.y, pos = blockIdx.x, tid = threadIdx.x;
  const long sbase = (long)bb * nSorted;
  __shared__ float wgt[NHASH];
  __shared__ long jrow[NHASH];
  if (tid < NHASH) {
    long j = undo[sbase + (long)tid * Tt + pos];
    jrow[tid] = j;
    wgt[tid] = sl[sbase + j];
  }
  __syncthreads();
  if (tid == 0) {
    float mx = -3.4e38f;
    for (int h = 0; h < NHASH; ++h) mx = fmaxf(mx, wgt[h]);
    float s = 0.f;
    for (int h = 0; h < NHASH; ++h) { wgt[h] = expf(wgt[h] - mx); s += wgt[h]; }
    for (int h = 0; h < NHASH; ++h) wgt[h] /= s;
  }
  __syncthreads();
  for (int dcol = tid; dcol < dv; dcol += blockDim.x) {
    float a = 0.f;
    for (int h = 0; h < NHASH; ++h) a += wgt[h] * so[(sbase + jrow[h]) * dv + dcol];
    outp[(long)bb * outBatchStride + (long)pos * dv + dcol] = a;
  }
}

// ---------------------------------------------------------------------------
// Copy x embedding into X2[:, 1024:2048, :] via ASYNCcnt-tracked LDS bounce
// (global_load_async_to_lds / global_store_async_from_lds, ISA §15.18.3).
// Low 32 bits of a generic __shared__ pointer are the LDS byte offset.
// ---------------------------------------------------------------------------
__global__ void k_concat_async(const float* __restrict__ src, float* __restrict__ dst) {
  __shared__ float buf[256];
  const int tid = threadIdx.x;
  long idx = (long)blockIdx.x * 256 + tid;  // over 8*1024*512 elements
  long b = idx >> 19;
  long rem = idx & ((1L << 19) - 1);
  long t = rem >> 9;
  int dcol = (int)(rem & 511);
  unsigned lds = (unsigned)(uintptr_t)&buf[tid];
  unsigned long long sa = (unsigned long long)(uintptr_t)(src + idx);
  unsigned long long da =
      (unsigned long long)(uintptr_t)(dst + ((((b << 11) + 1024 + t) << 9) + dcol));
  asm volatile("global_load_async_to_lds_b32 %0, %1, off" ::"v"(lds), "v"(sa)
               : "memory");
  asm volatile("s_wait_asynccnt 0x0" ::: "memory");
  asm volatile("global_store_async_from_lds_b32 %0, %1, off" ::"v"(da), "v"(lds)
               : "memory");
  asm volatile("s_wait_asynccnt 0x0" ::: "memory");
}

// ---------------------------------------------------------------------------
// Persistent GRU: one 32-wave workgroup per sequence; gh = h @ Whh^T via WMMA
// (M=16, rows 8..15 zero); h fp32 in registers, f16 mirror in LDS.
// Block 0: x-GRU (2048 steps); block 1: y-GRU (1024 steps).
// ---------------------------------------------------------------------------
__global__ void __launch_bounds__(1024) k_gru(
    const float* __restrict__ gx0, float* __restrict__ out0, int T0,
    const float* __restrict__ gx1, float* __restrict__ out1, int T1,
    const float* __restrict__ whh, const float* __restrict__ bhh) {
  __shared__ _Float16 hf[8 * 512];
  __shared__ float ghs[8 * 1536];
  const float* gx = (blockIdx.x == 0) ? gx0 : gx1;
  float* outp = (blockIdx.x == 0) ? out0 : out1;
  const int Tt = (blockIdx.x == 0) ? T0 : T1;
  const int tid = threadIdx.x;
  const int wave = tid >> 5;
  const int l = tid & 31, lm = l & 15, lh = l >> 4;
  const int D = 512, D3 = 1536;
  float hreg[4] = {0.f, 0.f, 0.f, 0.f};
#pragma unroll
  for (int e = 0; e < 4; ++e) hf[tid + e * 1024] = (_Float16)0.0f;
  __syncthreads();

  for (int step = 0; step < Tt; ++step) {
    for (int nt = wave; nt < (D3 >> 4); nt += 32) {  // 96 N-tiles / 32 waves
      const int col0 = nt << 4;
      v8f acc = {};
      for (int k0 = 0; k0 < D; k0 += 32) {
        v16h af = {};
        if (lm < 8) {  // rows 8..15 are zero padding
          const v8h a0 = *(const v8h*)&hf[lm * 512 + k0 + lh * 16];
          const v8h a1 = *(const v8h*)&hf[lm * 512 + k0 + lh * 16 + 8];
          af = CAT16(a0, a1);
        }
        const float4* bp = (const float4*)(whh + (long)(col0 + lm) * D + k0 + lh * 16);
        acc = wmma_f32_f16(af, pack16(bp[0], bp[1], bp[2], bp[3]), acc);
      }
      store_c_frag(acc, [&](int m, int n, float val) {
        if (m < 8) ghs[m * D3 + col0 + n] = val;
      });
    }
    __syncthreads();
#pragma unroll
    for (int e = 0; e < 4; ++e) {  // gate math for the 8x512 hidden state
      int i = tid + e * 1024;
      int b = i >> 9, dcol = i & 511;
      long base = ((long)b * Tt + step) * D3;
      float xr = gx[base + dcol];
      float xz = gx[base + D + dcol];
      float xn = gx[base + 2 * D + dcol];
      float hr = ghs[b * D3 + dcol] + bhh[dcol];
      float hz = ghs[b * D3 + D + dcol] + bhh[D + dcol];
      float hn = ghs[b * D3 + 2 * D + dcol] + bhh[2 * D + dcol];
      float r = sigmf(xr + hr);
      float z = sigmf(xz + hz);
      float nn = tanhf(xn + r * hn);
      float h = (1.0f - z) * nn + z * hreg[e];
      hreg[e] = h;
      outp[((long)b * Tt + step) * D + dcol] = h;
      hf[b * 512 + dcol] = (_Float16)h;
    }
    __syncthreads();
  }
}

// ---------------------------------------------------------------------------
// LayerNorm per row (D = 512)
// ---------------------------------------------------------------------------
__global__ void k_layernorm(const float* __restrict__ x, const float* __restrict__ g,
                            const float* __restrict__ bb, float* __restrict__ outp) {
  __shared__ float red[256];
  __shared__ float red2[256];
  const long row = blockIdx.x;
  const int tid = threadIdx.x;
  const float* xr = x + row * 512;
  float s = 0.f, s2 = 0.f;
  for (int k = tid; k < 512; k += 256) { float v = xr[k]; s += v; s2 += v * v; }
  red[tid] = s; red2[tid] = s2;
  __syncthreads();
  for (int st = 128; st > 0; st >>= 1) {
    if (tid < st) { red[tid] += red[tid + st]; red2[tid] += red2[tid + st]; }
    __syncthreads();
  }
  float mean = red[0] * (1.0f / 512.0f);
  float var = red2[0] * (1.0f / 512.0f) - mean * mean;
  float inv = rsqrtf(var + 1e-7f);
  for (int k = tid; k < 512; k += 256)
    outp[row * 512 + k] = (xr[k] - mean) * inv * g[k] + bb[k];
}

// ---------------------------------------------------------------------------
// Output masks (bool -> 0/1 float)
// ---------------------------------------------------------------------------
__global__ void k_masks(const int* __restrict__ xs, const int* __restrict__ refs,
                        float* __restrict__ ctx, float* __restrict__ tgt) {
  int idx = blockIdx.x * blockDim.x + threadIdx.x;
  if (idx < 8 * 2048) {
    int b = idx >> 11, j = idx & 2047;
    int v = (j < 1024) ? refs[b * 1024 + j] : xs[b * 1024 + (j - 1024)];
    ctx[idx] = (v == PADTOK) ? 1.0f : 0.0f;
  }
  if (idx < 8 * 1024) tgt[idx] = (refs[idx] == PADTOK) ? 1.0f : 0.0f;
}

// ---------------------------------------------------------------------------
extern "C" void kernel_launch(void* const* d_in, const int* in_sizes, int n_in,
                              void* d_out, int out_size, void* d_ws, size_t ws_size,
                              hipStream_t stream) {
  (void)in_sizes; (void)n_in; (void)out_size; (void)ws_size;
  const int* xs = (const int*)d_in[0];
  const int* ys = (const int*)d_in[1];
  const int* refs = (const int*)d_in[2];
  const float* emb = (const float*)d_in[3];
  const float* w_ih = (const float*)d_in[4];
  const float* w_hh = (const float*)d_in[5];
  const float* b_ih = (const float*)d_in[6];
  const float* b_hh = (const float*)d_in[7];
  const float* Wqk = (const float*)d_in[8];
  const float* Wv = (const float*)d_in[9];
  const float* Wout = (const float*)d_in[10];
  const float* bout = (const float*)d_in[11];
  const float* ln_g = (const float*)d_in[12];
  const float* ln_b = (const float*)d_in[13];
  float* out = (float*)d_out;

  const long Bn = 8, T = 1024, T2 = 2048, D = 512, H = 8, DH = 64, L = 6;

  // ---- workspace layout (bump allocator; SO region phase-aliased) ----
  size_t off = 0;
  char* wsb = (char*)d_ws;
  auto alloc = [&](size_t bytes) {
    size_t o = off;
    off += (bytes + 255) & ~(size_t)255;
    return (void*)(wsb + o);
  };
  float* XE = (float*)alloc(Bn * T * D * 4);
  float* YE = (float*)alloc(Bn * T * D * 4);
  float* RE = (float*)alloc(Bn * T * D * 4);
  float* XA = (float*)alloc(Bn * T2 * D * 4);  // X2 / encoder ping
  float* XB = (float*)alloc(Bn * T2 * D * 4);  // encoder pong
  float* QKh = (float*)alloc(Bn * H * T2 * DH * 4);
  float* Vh = (float*)alloc(Bn * H * T2 * DH * 4);
  float* ROTM = (float*)alloc(512 * 256 * 4);
  int* BUK = (int*)alloc(64L * 16384 * 4);
  int* STK = (int*)alloc(64L * 16384 * 4);
  int* UND = (int*)alloc(64L * 16384 * 4);
  int* CNT = (int*)alloc(64L * 512 * 4);
  float* SL = (float*)alloc(64L * 16384 * 4);
  float* OH = (float*)alloc(Bn * H * T2 * DH * 4);
  float* OB = (float*)alloc(Bn * T2 * D * 4);
  float* SO = (float*)alloc(268435456);  // sorted attn out (max, phase-aliased)
  float* ROT = SO;                       // alias: dead before SO is written
  float* GX = SO;                        // alias: GRU phase only
  float* GY = (float*)((char*)SO + 134217728);

  dim3 blk256(256);
  long nemb = Bn * T * D;
  int gemb = (int)((nemb + 255) / 256);
  k_embed<<<gemb, blk256, 0, stream>>>(xs, emb, XE, nemb);
  k_embed<<<gemb, blk256, 0, stream>>>(ys, emb, YE, nemb);
  k_embed<<<gemb, blk256, 0, stream>>>(refs, emb, RE, nemb);

  // ---- first LSH attention: qk = x emb (d=512), v = ref emb ----
  {
    const int R = 16, NB = 8, Tt = 1024, nS = NHASH * Tt, nchunks = nS / 32;
    long nrot = 512L * NHASH * R;  // [512,128]
    k_randn<<<(int)((nrot + 255) / 256), blk256, 0, stream>>>(ROTM, nrot, 0xC0FFEE01u);
    k_gemm<<<dim3(128 / 64, 8192 / 64), blk256, 0, stream>>>(
        XE, ROTM, nullptr, ROT, 8192, 128, 512, 0, 128, 0, 0);
    long tb = (long)NB * NHASH * Tt;
    k_buckets<<<(int)((tb + 255) / 256), blk256, 0, stream>>>(ROT, BUK, tb, 10, R);
    k_sort<<<1, 64, 0, stream>>>(BUK, STK, UND, CNT, NB, nS, NHASH * 2 * R);
    k_lsh_attn<<<dim3(nchunks, NB), blk256, 0, stream>>>(
        XE, RE, STK, xs, SO, SL, Tt, Tt - 1, nchunks - 1, 0, 512, 512, nS,
        0.044194174f);
    k_combine<<<dim3(Tt, NB), 128, 0, stream>>>(SO, SL, UND, XA, Tt, 512, nS,
                                                (long)T2 * D);
    k_concat_async<<<gemb, blk256, 0, stream>>>(XE, XA);
  }

  // ---- GRU input projections + persistent recurrent kernel ----
  k_gemm<<<dim3(1536 / 64, 16384 / 64), blk256, 0, stream>>>(
      XA, w_ih, b_ih, GX, 16384, 1536, 512, 1, 512, 0, 0);
  k_gemm<<<dim3(1536 / 64, 8192 / 64), blk256, 0, stream>>>(
      YE, w_ih, b_ih, GY, 8192, 1536, 512, 1, 512, 0, 0);
  float* YOUT = out + Bn * T2 * D;  // y output region of d_out
  k_gru<<<2, 1024, 0, stream>>>(GX, XA, (int)T2, GY, YOUT, (int)T, w_hh, b_hh);

  // ---- 6 light-encoder layers ----
  float* cur = XA;
  for (int l = 0; l < (int)L; ++l) {
    const float* wqk = Wqk + (long)l * D * D;
    const float* wv = Wv + (long)l * D * D;
    const float* wo = Wout + (long)l * D * D;
    const float* bo = bout + (long)l * D;
    const float* lg = ln_g + (long)l * D;
    const float* lb = ln_b + (long)l * D;
    // shared-QK and V projections, head-split outputs [64, 2048, 64]
    k_gemm<<<dim3(512 / 64, 16384 / 64), blk256, 0, stream>>>(
        cur, wqk, nullptr, QKh, 16384, 512, 512, 0, 512, 1, 0);
    k_gemm<<<dim3(512 / 64, 16384 / 64), blk256, 0, stream>>>(
        cur, wv, nullptr, Vh, 16384, 512, 512, 0, 512, 1, 0);
    const int R = 32, NB = 64, Tt = 2048, nS = NHASH * Tt, nchunks = nS / 32;
    long nrot = 64L * NHASH * R;  // [64,256]
    k_randn<<<(int)((nrot + 255) / 256), blk256, 0, stream>>>(ROTM, nrot,
                                                             0xABCD0000u + l);
    k_gemm<<<dim3(256 / 64, 131072 / 64), blk256, 0, stream>>>(
        QKh, ROTM, nullptr, ROT, 131072, 256, 64, 0, 256, 0, 0);
    long tb = (long)NB * NHASH * Tt;
    k_buckets<<<(int)((tb + 255) / 256), blk256, 0, stream>>>(ROT, BUK, tb, 11, R);
    k_sort<<<1, 64, 0, stream>>>(BUK, STK, UND, CNT, NB, nS, NHASH * 2 * R);
    k_lsh_attn<<<dim3(nchunks, NB), blk256, 0, stream>>>(
        QKh, Vh, STK, xs, SO, SL, Tt, Tt - 1, nchunks - 1, 3, 64, 64, nS, 0.125f);
    k_combine<<<dim3(Tt, NB), 128, 0, stream>>>(SO, SL, UND, OH, Tt, 64, nS,
                                                (long)Tt * 64);
    // out projection (gathers head-split OH) + bias
    k_gemm<<<dim3(512 / 64, 16384 / 64), blk256, 0, stream>>>(
        OH, wo, bo, OB, 16384, 512, 512, 0, 512, 0, 1);
    float* nxt = (l == (int)L - 1) ? out : ((cur == XA) ? XB : XA);
    k_layernorm<<<16384, 256, 0, stream>>>(OB, lg, lb, nxt);
    cur = nxt;
  }

  // ---- boolean mask outputs ----
  float* CTX = out + Bn * T2 * D + Bn * T * D;
  float* TGT = CTX + Bn * T2;
  k_masks<<<(16384 + 255) / 256, blk256, 0, stream>>>(xs, refs, CTX, TGT);
}
